// NonLocal2d_38843684225269
// MI455X (gfx1250) — compile-verified, matrix operand order checked by reading, not run on hardware
//
#include <hip/hip_runtime.h>

// ---------------------------------------------------------------------------
// NonLocal2d block for MI455X (gfx1250, wave32, WMMA f16 -> f32 acc).
//
// ~155 GFLOP vs ~0.5 GB HBM -> mildly compute-bound at f16 WMMA rate.
// All GEMMs: v_wmma_f32_16x16x32_f16, 128x128 workgroup tiles, double-
// buffered LDS (1 barrier / k-step), f16 B operands column-major so staging
// is pure 16B copies. Tile staging uses the gfx1250 async global->LDS path
// (GLOBAL_LOAD_ASYNC_TO_LDS_B128 + s_wait_asynccnt), which co-issues with
// the WMMA pipe and skips the VGPR round-trip; the f32 input x (needs
// f32->f16 convert) keeps the register path.
// ---------------------------------------------------------------------------

typedef __attribute__((ext_vector_type(16))) _Float16 v16h;
typedef __attribute__((ext_vector_type(8)))  _Float16 v8h;
typedef __attribute__((ext_vector_type(4)))  _Float16 v4h;
typedef __attribute__((ext_vector_type(8)))  float    v8f;

#if defined(__has_builtin)
#  if __has_builtin(__builtin_amdgcn_global_load_async_to_lds_b128) && \
      __has_builtin(__builtin_amdgcn_s_wait_asynccnt)
#    define USE_ASYNC_LDS 1
#  endif
#endif
#ifndef USE_ASYNC_LDS
#  define USE_ASYNC_LDS 0
#endif

#if USE_ASYNC_LDS
// builtin signature (from clang diagnostic): int4 AS1* src, int4 AS3* dst, imm, imm
typedef __attribute__((__vector_size__(4 * sizeof(int)))) int v4i;
typedef __attribute__((address_space(1))) v4i gv4i;   // global int4
typedef __attribute__((address_space(3))) v4i lv4i;   // LDS int4
#endif

#define LSTR 40  // padded LDS row stride (halfs): 80B rows keep b128 16B-aligned

__device__ __forceinline__ v16h frag16(const _Float16* p) {
    // ISA 16-bit operand layout: elements 0..7 = k koff..koff+7,
    // elements 8..15 = k koff+16..koff+23  (two 16B ds_load_b128)
    v8h lo = *(const v8h*)p;
    v8h hi = *(const v8h*)(p + 16);
    return __builtin_shufflevector(lo, hi, 0,1,2,3,4,5,6,7,8,9,10,11,12,13,14,15);
}

// BM=BN=128, BK=32. 256 threads = 8 waves, each wave -> 32x64 output
// (2 A-frags x 4 B-frags = 8 wmma per k-step).
template<bool B_F32, bool OUT_F16, bool TRANS, bool BIAS, bool RES>
__global__ __launch_bounds__(256) void gemm_wmma_kernel(
    const _Float16* __restrict__ A, unsigned long long sA,   // batch strides (elems)
    const void*     __restrict__ B, unsigned long long sB,
    void*           __restrict__ C, unsigned long long sC,
    const float*    __restrict__ bias,
    const float*    __restrict__ res, unsigned long long sRes,
    int K, int lda, int ldb, int ldc, float scale)
{
    __shared__ __align__(16) _Float16 ldsA[2][128 * LSTR];
    __shared__ __align__(16) _Float16 ldsB[2][128 * LSTR];   // [n][k]

    const int tid  = threadIdx.x;
    const int lane = tid & 31;
    const int wv   = tid >> 5;
    const int m0   = blockIdx.y * 128;
    const int n0   = blockIdx.x * 128;
    const int z    = blockIdx.z;

    const _Float16* Ab = A + (size_t)z * sA;
    const char*     Bb = (const char*)B + (size_t)z * sB * (B_F32 ? 4 : 2);

    // wave -> 32x64 region: 4 row-blocks x 2 col-blocks covers 128x128
    const int wr   = wv & 3;
    const int wc   = wv >> 2;
    const int koff = (lane >> 4) << 3;
    const int mA0  = wr * 32 + (lane & 15);
    const int nB0  = wc * 64 + (lane & 15);

    // staging split (256 threads)
    const int aRow  = tid >> 1, aKg = (tid & 1) << 4;  // A: 128 rows x 32 k
    const int bnRow = tid >> 1, bKg = (tid & 1) << 4;  // B col-major: 128 n x 32 k
    const int bkRow = tid >> 3, bNg = (tid & 7) << 4;  // B f32 row-major: 32 k x 128 n

    v8h ra0, ra1, rb0, rb1;
    float4 rf0, rf1, rf2, rf3;

    auto gloadA = [&](int k0) {
        const _Float16* srcA = Ab + (size_t)(m0 + aRow) * lda + (k0 + aKg);
        ra0 = *(const v8h*)srcA;
        ra1 = *(const v8h*)(srcA + 8);
        __builtin_prefetch(srcA + 32, 0, 1);
    };
    auto lstoreA = [&](int buf) {
        *(v8h*)&ldsA[buf][aRow * LSTR + aKg]     = ra0;
        *(v8h*)&ldsA[buf][aRow * LSTR + aKg + 8] = ra1;
    };
    auto gloadB = [&](int k0) {
        if constexpr (B_F32) {
            const float* srcB = (const float*)Bb + (size_t)(k0 + bkRow) * ldb + (n0 + bNg);
            rf0 = ((const float4*)srcB)[0];
            rf1 = ((const float4*)srcB)[1];
            rf2 = ((const float4*)srcB)[2];
            rf3 = ((const float4*)srcB)[3];
            __builtin_prefetch(srcB + (size_t)32 * ldb, 0, 1);
        } else {
            const _Float16* srcB = (const _Float16*)Bb + (size_t)(n0 + bnRow) * ldb + (k0 + bKg);
            rb0 = *(const v8h*)srcB;
            rb1 = *(const v8h*)(srcB + 8);
            __builtin_prefetch(srcB + 32, 0, 1);
        }
    };
    auto lstoreB = [&](int buf) {
        if constexpr (B_F32) {
            _Float16 h[16] = {
                (_Float16)rf0.x, (_Float16)rf0.y, (_Float16)rf0.z, (_Float16)rf0.w,
                (_Float16)rf1.x, (_Float16)rf1.y, (_Float16)rf1.z, (_Float16)rf1.w,
                (_Float16)rf2.x, (_Float16)rf2.y, (_Float16)rf2.z, (_Float16)rf2.w,
                (_Float16)rf3.x, (_Float16)rf3.y, (_Float16)rf3.z, (_Float16)rf3.w };
            #pragma unroll
            for (int j = 0; j < 16; ++j)
                ldsB[buf][(bNg + j) * LSTR + bkRow] = h[j];
        } else {
            *(v8h*)&ldsB[buf][bnRow * LSTR + bKg]     = rb0;
            *(v8h*)&ldsB[buf][bnRow * LSTR + bKg + 8] = rb1;
        }
    };

#if USE_ASYNC_LDS
    // async global->LDS DMA for the pure-copy operands (A always; B when f16)
    auto asyncAB = [&](int k0, int buf) {
        const _Float16* sA = Ab + (size_t)(m0 + aRow) * lda + (k0 + aKg);
        _Float16* dA = &ldsA[buf][aRow * LSTR + aKg];
        __builtin_amdgcn_global_load_async_to_lds_b128((gv4i*)sA, (lv4i*)dA, 0, 0);
        __builtin_amdgcn_global_load_async_to_lds_b128((gv4i*)sA, (lv4i*)dA, 16, 0);
        if constexpr (!B_F32) {
            const _Float16* sB = (const _Float16*)Bb + (size_t)(n0 + bnRow) * ldb + (k0 + bKg);
            _Float16* dB = &ldsB[buf][bnRow * LSTR + bKg];
            __builtin_amdgcn_global_load_async_to_lds_b128((gv4i*)sB, (lv4i*)dB, 0, 0);
            __builtin_amdgcn_global_load_async_to_lds_b128((gv4i*)sB, (lv4i*)dB, 16, 0);
        }
    };
#endif

    v8f acc[2][4] = {};
    const int nk = K >> 5;

#if USE_ASYNC_LDS
    asyncAB(0, 0);
    if constexpr (B_F32) { gloadB(0); lstoreB(0); }
    __builtin_amdgcn_s_wait_asynccnt(0);
    __syncthreads();
#else
    gloadA(0); gloadB(0);
    lstoreA(0); lstoreB(0);
    __syncthreads();
#endif

    int cur = 0;
    for (int kt = 0; kt < nk; ++kt) {
        const bool hasNext = (kt + 1 < nk);
        if (hasNext) {   // stage tile kt+1 while tile kt computes
#if USE_ASYNC_LDS
            asyncAB((kt + 1) << 5, cur ^ 1);
            if constexpr (B_F32) gloadB((kt + 1) << 5);
#else
            gloadA((kt + 1) << 5);
            gloadB((kt + 1) << 5);
#endif
        }

        v16h af[2], bf[4];
        af[0] = frag16(&ldsA[cur][mA0 * LSTR + koff]);
        af[1] = frag16(&ldsA[cur][(mA0 + 16) * LSTR + koff]);
        #pragma unroll
        for (int j = 0; j < 4; ++j)
            bf[j] = frag16(&ldsB[cur][(nB0 + 16 * j) * LSTR + koff]);

        #pragma unroll
        for (int i = 0; i < 2; ++i)
            #pragma unroll
            for (int j = 0; j < 4; ++j)
                acc[i][j] = __builtin_amdgcn_wmma_f32_16x16x32_f16(
                    false, af[i], false, bf[j], (short)0, acc[i][j], false, false);

        if (hasNext) {
#if USE_ASYNC_LDS
            if constexpr (B_F32) lstoreB(cur ^ 1);
            __builtin_amdgcn_s_wait_asynccnt(0);
#else
            lstoreA(cur ^ 1);
            lstoreB(cur ^ 1);
#endif
            __syncthreads();
            cur ^= 1;
        }
    }

    // epilogue: D layout lane->N, vgpr r -> M = base + r + 8*(lane>=16)
    char* Cb = (char*)C + (size_t)z * sC * (OUT_F16 ? 2 : 4);
    const float* resb = RES ? (res + (size_t)z * sRes) : nullptr;
    const int nbase = n0 + wc * 64 + (lane & 15);
    const int mbase = m0 + wr * 32 + ((lane >> 4) << 3);
    #pragma unroll
    for (int i = 0; i < 2; ++i) {
        #pragma unroll
        for (int j = 0; j < 4; ++j) {
            const int n = nbase + j * 16;
            #pragma unroll
            for (int r = 0; r < 8; ++r) {
                const int m = mbase + i * 16 + r;
                float v = acc[i][j][r] * scale;
                if (BIAS) v += bias[m];
                if (RES)  v += resb[(size_t)m * ldc + n];
                const size_t idx = TRANS ? ((size_t)n * ldc + m) : ((size_t)m * ldc + n);
                if (OUT_F16) ((_Float16*)Cb)[idx] = (_Float16)v;
                else         ((float*)Cb)[idx]    = v;
            }
        }
    }
}

// ---- f32 -> f16 (vectorized x4) ----
__global__ __launch_bounds__(256) void f32_to_f16_kernel(
    const float* __restrict__ in, _Float16* __restrict__ out, int n4)
{
    int i = blockIdx.x * 256 + threadIdx.x;
    if (i >= n4) return;
    float4 v = ((const float4*)in)[i];
    v4h h; h[0] = (_Float16)v.x; h[1] = (_Float16)v.y; h[2] = (_Float16)v.z; h[3] = (_Float16)v.w;
    *((v4h*)out + i) = h;
}

// ---- maxpool2x2: x f32 [b][c][64][64] -> pool_t f16 [b][m=32*32][c] ----
__global__ __launch_bounds__(256) void pool2x2_f16_kernel(
    const float* __restrict__ x, _Float16* __restrict__ out)
{
    int i  = blockIdx.x * 256 + threadIdx.x;   // over [b][c][hw_out]
    int hw = i & 1023;
    int bc = i >> 10;
    int b  = bc >> 10;
    int c  = bc & 1023;
    int oh = hw >> 5, ow = hw & 31;
    const float* src = x + ((size_t)bc << 12) + (size_t)(oh * 2) * 64 + ow * 2;
    float m = fmaxf(fmaxf(src[0], src[1]), fmaxf(src[64], src[65]));
    out[((size_t)b * 1024 + hw) * 1024 + c] = (_Float16)m;   // spatial-major
}

// ---- row softmax: 1024 f32 -> 1024 f16 in-place at row start ----
__global__ __launch_bounds__(256) void softmax_rows_kernel(float* __restrict__ logits)
{
    __shared__ float red[256];
    const int t = threadIdx.x;
    float* rp = logits + (size_t)blockIdx.x * 1024;
    float4 v = *(const float4*)(rp + t * 4);
    float mx = fmaxf(fmaxf(v.x, v.y), fmaxf(v.z, v.w));
    red[t] = mx; __syncthreads();
    #pragma unroll
    for (int s = 128; s > 0; s >>= 1) { if (t < s) red[t] = fmaxf(red[t], red[t + s]); __syncthreads(); }
    const float rowmax = red[0];
    __syncthreads();
    float e0 = __expf(v.x - rowmax), e1 = __expf(v.y - rowmax);
    float e2 = __expf(v.z - rowmax), e3 = __expf(v.w - rowmax);
    red[t] = e0 + e1 + e2 + e3; __syncthreads();
    #pragma unroll
    for (int s = 128; s > 0; s >>= 1) { if (t < s) red[t] += red[t + s]; __syncthreads(); }
    const float inv = 1.0f / red[0];
    // all reads of this row precede the first barrier -> in-place f16 is safe
    v4h h; h[0] = (_Float16)(e0 * inv); h[1] = (_Float16)(e1 * inv);
           h[2] = (_Float16)(e2 * inv); h[3] = (_Float16)(e3 * inv);
    *(v4h*)((_Float16*)rp + t * 4) = h;
}

extern "C" void kernel_launch(void* const* d_in, const int* in_sizes, int n_in,
                              void* d_out, int out_size, void* d_ws, size_t ws_size,
                              hipStream_t stream) {
    (void)in_sizes; (void)n_in; (void)out_size; (void)ws_size;
    const float* x       = (const float*)d_in[0];
    const float* theta_w = (const float*)d_in[1];
    const float* theta_b = (const float*)d_in[2];
    const float* phi_w   = (const float*)d_in[3];
    const float* phi_b   = (const float*)d_in[4];
    const float* g_w     = (const float*)d_in[5];
    const float* g_b     = (const float*)d_in[6];
    const float* out_w   = (const float*)d_in[7];
    const float* out_b   = (const float*)d_in[8];
    float* y = (float*)d_out;

    constexpr int Bn = 8, Cin = 1024, Di = 512, Cout = 1024;
    constexpr int HW = 4096, Np = 1024;

    char* ws = (char*)d_ws;
    size_t off = 0;
    auto alloc = [&](size_t bytes) -> char* {
        char* p = ws + off; off += (bytes + 255) & ~(size_t)255; return p;
    };
    _Float16* pool_t = (_Float16*)alloc((size_t)Bn * Np * Cin * 2);   // [b][m][c]
    _Float16* thetaT = (_Float16*)alloc((size_t)Bn * HW * Di * 2);    // [b][n][d]
    _Float16* phiT   = (_Float16*)alloc((size_t)Bn * Np * Di * 2);    // [b][m][d]
    _Float16* g_h    = (_Float16*)alloc((size_t)Bn * Di * Np * 2);    // [b][d][m]
    _Float16* t_xT   = (_Float16*)alloc((size_t)Bn * HW * Di * 2);    // [b][n][d]
    _Float16* wt_h   = (_Float16*)alloc((size_t)Di * Cin * 2);
    _Float16* wp_h   = (_Float16*)alloc((size_t)Di * Cin * 2);
    _Float16* wg_h   = (_Float16*)alloc((size_t)Di * Cin * 2);
    _Float16* wo_h   = (_Float16*)alloc((size_t)Cout * Di * 2);
    float*    logits = (float*)alloc((size_t)HW * Np * 4);            // per-batch reuse

    // 1) weights -> f16
    f32_to_f16_kernel<<<dim3((Di * Cin / 4) / 256), 256, 0, stream>>>(theta_w, wt_h, Di * Cin / 4);
    f32_to_f16_kernel<<<dim3((Di * Cin / 4) / 256), 256, 0, stream>>>(phi_w,   wp_h, Di * Cin / 4);
    f32_to_f16_kernel<<<dim3((Di * Cin / 4) / 256), 256, 0, stream>>>(g_w,     wg_h, Di * Cin / 4);
    f32_to_f16_kernel<<<dim3((Cout * Di / 4) / 256), 256, 0, stream>>>(out_w,  wo_h, Cout * Di / 4);

    // 2) maxpool2x2 -> f16 spatial-major
    pool2x2_f16_kernel<<<dim3((Bn * Cin * Np) / 256), 256, 0, stream>>>(x, pool_t);

    // 3) theta: w[512x1024] x x_f32[1024x4096] -> thetaT[b][n][d] (TRANS, +bias)
    gemm_wmma_kernel<true, true, true, true, false>
        <<<dim3(HW / 128, Di / 128, Bn), 256, 0, stream>>>(
        wt_h, 0ull, x, (unsigned long long)Cin * HW, thetaT, (unsigned long long)HW * Di,
        theta_b, nullptr, 0ull, Cin, Cin, HW, Di, 1.0f);

    // 4) phi -> phiT[b][m][d] (TRANS, +bias); g -> g_h[b][d][m] (+bias)
    gemm_wmma_kernel<false, true, true, true, false>
        <<<dim3(Np / 128, Di / 128, Bn), 256, 0, stream>>>(
        wp_h, 0ull, pool_t, (unsigned long long)Np * Cin, phiT, (unsigned long long)Np * Di,
        phi_b, nullptr, 0ull, Cin, Cin, Cin, Di, 1.0f);
    gemm_wmma_kernel<false, true, false, true, false>
        <<<dim3(Np / 128, Di / 128, Bn), 256, 0, stream>>>(
        wg_h, 0ull, pool_t, (unsigned long long)Np * Cin, g_h, (unsigned long long)Di * Np,
        g_b, nullptr, 0ull, Cin, Cin, Cin, Np, 1.0f);

    // 5) attention, per batch (reuses one 16 MiB logits buffer)
    const float scale = 0.04419417382415922f;  // 512^-0.5
    for (int b = 0; b < Bn; ++b) {
        // logits[n][m] = scale * thetaT[b] @ phiT[b]^T
        gemm_wmma_kernel<false, false, false, false, false>
            <<<dim3(Np / 128, HW / 128, 1), 256, 0, stream>>>(
            thetaT + (size_t)b * HW * Di, 0ull, phiT + (size_t)b * Np * Di, 0ull,
            logits, 0ull, nullptr, nullptr, 0ull, Di, Di, Di, Np, scale);
        // softmax rows -> f16 p in-place (rows stay 4096B apart -> ldb = 2048 halfs)
        softmax_rows_kernel<<<dim3(HW), 256, 0, stream>>>(logits);
        // t_xT[n][d] = (g[b] @ p^T)^T   (TRANS store gives col-major B for GEMM 6)
        gemm_wmma_kernel<false, true, true, false, false>
            <<<dim3(HW / 128, Di / 128, 1), 256, 0, stream>>>(
            g_h + (size_t)b * Di * Np, 0ull, logits, 0ull,
            t_xT + (size_t)b * HW * Di, 0ull, nullptr, nullptr, 0ull,
            Np, Np, 2048, Di, 1.0f);
    }

    // 6) out: w[1024x512] x t[512x4096] + out_b + x -> y (f32)
    gemm_wmma_kernel<false, false, false, true, true>
        <<<dim3(HW / 128, Cout / 128, Bn), 256, 0, stream>>>(
        wo_h, 0ull, t_xT, (unsigned long long)HW * Di, y, (unsigned long long)Cout * HW,
        out_b, x, (unsigned long long)Cin * HW, Di, Di, Di, HW, 1.0f);
}
